// HigherOrderGNN_10557029614296
// MI455X (gfx1250) — matrix operand order repaired and checked by reading, not compiled, not run on hardware
//
#include <hip/hip_runtime.h>

// ---------------------------------------------------------------------------
// HigherOrderGNN on MI455X (gfx1250, wave32, WMMA, bf16 activations)
//   act = bf16(x)
//   agg = scatter_add(act)           (fp32 atomics, bf16 gathers)
//   act = PReLU( agg@Wrel1 + brel1 + act@Wroot1 )   (bf16, in place)
//   agg = scatter_add(act)           (re-zeroed, reused)
//   act = PReLU( agg@Wrel2 + brel2 + act@Wroot2 )   (bf16, in place)
//   out = segpool(act) @ Wpost + bpost
// GEMM: [N x 512] x [512 x 256] via v_wmma_f32_16x16x32_bf16,
// K = [agg(fp32, cvt in-reg) | act(bf16, direct)] concatenated.
// Each wave carries TWO 16-row A fragments (32 rows) against 8 column
// tiles, so every B fragment load feeds two WMMAs (1 load-pair : 2 wmma).
// B is pre-swizzled into per-lane fragment order and is L2-resident.
// ---------------------------------------------------------------------------

typedef __attribute__((ext_vector_type(16))) __bf16 v16bf;
typedef __attribute__((ext_vector_type(8)))  float  v8f;

#define NN   100000
#define EE   320000
#define HH   256
#define GG   64
#define KTOT 512

#define WMMA_BF16(A, B, C) \
    __builtin_amdgcn_wmma_f32_16x16x32_bf16(false, (A), false, (B), (short)0, (C), false, false)

union BFrag { v16bf v; uint4 q[2]; };

// Build a bf16 A fragment from fp32 memory (rows of agg).
// Per lane: K = g*8 + 0..7  and  K = 16 + g*8 + 0..7.
static __device__ inline BFrag load_a_f32(const float* __restrict__ base)
{
    float4 f0 = *(const float4*)(base);
    float4 f1 = *(const float4*)(base + 4);
    float4 f2 = *(const float4*)(base + 16);
    float4 f3 = *(const float4*)(base + 20);
    BFrag a;
    a.v[0]  = (__bf16)f0.x; a.v[1]  = (__bf16)f0.y;
    a.v[2]  = (__bf16)f0.z; a.v[3]  = (__bf16)f0.w;
    a.v[4]  = (__bf16)f1.x; a.v[5]  = (__bf16)f1.y;
    a.v[6]  = (__bf16)f1.z; a.v[7]  = (__bf16)f1.w;
    a.v[8]  = (__bf16)f2.x; a.v[9]  = (__bf16)f2.y;
    a.v[10] = (__bf16)f2.z; a.v[11] = (__bf16)f2.w;
    a.v[12] = (__bf16)f3.x; a.v[13] = (__bf16)f3.y;
    a.v[14] = (__bf16)f3.z; a.v[15] = (__bf16)f3.w;
    return a;
}

// Build a bf16 A fragment from bf16 memory (rows of act).
static __device__ inline BFrag load_a_bf16(const __bf16* __restrict__ base)
{
    BFrag a;
    a.q[0] = *(const uint4*)(base);
    a.q[1] = *(const uint4*)(base + 16);
    return a;
}

// -------------------------------------------------------------------
// K0: convert + swizzle weights into per-lane WMMA B-fragment order.
// Fragment storage index: (((kb*16 + ct)*32 + lane)*16 + e), bf16.
//   K = kb*32 + (lane>>4)*8 + (e&7) + (e>>3)*16 ;  n = ct*16 + (lane&15)
//   K < 256 -> Wrel[K][n], else Wroot[K-256][n]
// -------------------------------------------------------------------
__global__ void gnn_swizzle_weights(const float* __restrict__ Wrel,
                                    const float* __restrict__ Wroot,
                                    __bf16* __restrict__ Wswz)
{
    int idx  = blockIdx.x * 256 + threadIdx.x;      // 0 .. 512*256-1
    int e    = idx & 15;
    int lane = (idx >> 4) & 31;
    int ct   = (idx >> 9) & 15;
    int kb   = idx >> 13;
    int K    = kb * 32 + ((lane >> 4) * 8) + (e & 7) + ((e >> 3) * 16);
    int n    = ct * 16 + (lane & 15);
    float v  = (K < 256) ? Wrel[(size_t)K * HH + n]
                         : Wroot[(size_t)(K - 256) * HH + n];
    Wswz[idx] = (__bf16)v;
}

// -------------------------------------------------------------------
// K_cvt: act = bf16(x)   (one thread per 8 elements)
// -------------------------------------------------------------------
__global__ void gnn_cvt_bf16(const float* __restrict__ x,
                             __bf16* __restrict__ act)
{
    size_t i = ((size_t)blockIdx.x * 256 + threadIdx.x) * 8;
    if (i >= (size_t)NN * HH) return;
    float4 f0 = *(const float4*)(x + i);
    float4 f1 = *(const float4*)(x + i + 4);
    union { __bf16 h[8]; uint4 q; } p;
    p.h[0] = (__bf16)f0.x; p.h[1] = (__bf16)f0.y;
    p.h[2] = (__bf16)f0.z; p.h[3] = (__bf16)f0.w;
    p.h[4] = (__bf16)f1.x; p.h[5] = (__bf16)f1.y;
    p.h[6] = (__bf16)f1.z; p.h[7] = (__bf16)f1.w;
    *(uint4*)(act + i) = p.q;
}

// -------------------------------------------------------------------
// K1: edge scatter-add: agg[dst] += act[src]
// bf16 gathers (uint4 = 8 feats), fp32 global atomics.
// one thread per (edge, 8-column group)
// -------------------------------------------------------------------
__global__ void gnn_scatter_add(const __bf16* __restrict__ act,
                                const int* __restrict__ src,
                                const int* __restrict__ dst,
                                float* __restrict__ agg)
{
    int tid = blockIdx.x * 256 + threadIdx.x;
    int e   = tid >> 5;
    int c   = (tid & 31) << 3;
    if (e >= EE) return;
    int s = src[e];
    int d = dst[e];
    union { uint4 q; __bf16 h[8]; } v;
    v.q = *(const uint4*)(act + (size_t)s * HH + c);
    float* o = agg + (size_t)d * HH + c;
    #pragma unroll
    for (int j = 0; j < 8; ++j)
        atomicAdd(o + j, (float)v.h[j]);
}

// -------------------------------------------------------------------
// K2: fused GEMM + bias + PReLU.
//   out[r][:] = bf16( prelu( sum_k A[r][k] * Wswz[k][:] + bias ) )
//   A = [agg (fp32) | root (bf16)] along K (512).
//   64 threads = 2 waves per block; block covers 32 rows x 256 cols.
//   Wave w: rows blk*32 .. blk*32+31, col tiles ct = w*8 .. w*8+7.
//   Two A fragments per wave -> each B fragment feeds two WMMAs.
//   In-place safe (out may alias root): all root reads precede stores.
// -------------------------------------------------------------------
__global__ __launch_bounds__(64)
void gnn_gemm_bias_prelu(const float*  __restrict__ agg,
                         const __bf16* __restrict__ root,
                         const float*  __restrict__ bias,
                         const float*  __restrict__ slopePtr,
                         const __bf16* __restrict__ Wswz,
                         __bf16* out)
{
    const int lane   = threadIdx.x & 31;
    const int wave   = threadIdx.x >> 5;
    const int row0   = blockIdx.x * 32;
    const int g      = lane >> 4;   // lane half: A K-offset group / C M-offset
    const int arow0  = row0 + (lane & 15);
    const int arow1  = arow0 + 16;
    const int ctBase = wave * 8;

    v8f acc0[8], acc1[8];
    #pragma unroll
    for (int i = 0; i < 8; ++i) { acc0[i] = (v8f){}; acc1[i] = (v8f){}; }

    // ---- phase 1: K = 0..255 from agg (fp32 -> bf16 in-register) ----
    #pragma unroll
    for (int kb = 0; kb < 8; ++kb) {
        const int koff = kb * 32 + g * 8;
        BFrag a0 = load_a_f32(agg + (size_t)arow0 * HH + koff);
        BFrag a1 = load_a_f32(agg + (size_t)arow1 * HH + koff);

        if (kb == 7) {  // warm the bf16 root stream for phase 2
            __builtin_prefetch(root + (size_t)arow0 * HH + g * 8, 0, 1);
            __builtin_prefetch(root + (size_t)arow1 * HH + g * 8, 0, 1);
        }

        #pragma unroll
        for (int c8 = 0; c8 < 8; ++c8) {
            const __bf16* wb =
                Wswz + ((size_t)((kb * 16 + ctBase + c8) * 32 + lane)) * 16;
            BFrag b;
            b.q[0] = *(const uint4*)(wb);
            b.q[1] = *(const uint4*)(wb + 8);
            acc0[c8] = WMMA_BF16(a0.v, b.v, acc0[c8]);
            acc1[c8] = WMMA_BF16(a1.v, b.v, acc1[c8]);
        }
    }

    // ---- phase 2: K = 256..511 from root (bf16 direct) ----
    #pragma unroll
    for (int kb = 8; kb < 16; ++kb) {
        const int koff = (kb - 8) * 32 + g * 8;
        BFrag a0 = load_a_bf16(root + (size_t)arow0 * HH + koff);
        BFrag a1 = load_a_bf16(root + (size_t)arow1 * HH + koff);

        #pragma unroll
        for (int c8 = 0; c8 < 8; ++c8) {
            const __bf16* wb =
                Wswz + ((size_t)((kb * 16 + ctBase + c8) * 32 + lane)) * 16;
            BFrag b;
            b.q[0] = *(const uint4*)(wb);
            b.q[1] = *(const uint4*)(wb + 8);
            acc0[c8] = WMMA_BF16(a0.v, b.v, acc0[c8]);
            acc1[c8] = WMMA_BF16(a1.v, b.v, acc1[c8]);
        }
    }

    // ---- epilogue: bias + PReLU + bf16 store ----
    // C layout: VGPR i, lanes 0-15 -> M=i ; lanes 16-31 -> M=i+8 ; N = lane&15.
    const float slope = slopePtr[0];
    const int   r0    = row0 + g * 8;
    #pragma unroll
    for (int c8 = 0; c8 < 8; ++c8) {
        const int   col = (ctBase + c8) * 16 + (lane & 15);
        const float bv  = bias[col];
        #pragma unroll
        for (int i = 0; i < 8; ++i) {
            float v0 = acc0[c8][i] + bv;
            v0 = (v0 >= 0.0f) ? v0 : slope * v0;
            out[(size_t)(r0 + i) * HH + col] = (__bf16)v0;
            float v1 = acc1[c8][i] + bv;
            v1 = (v1 >= 0.0f) ? v1 : slope * v1;
            out[(size_t)(r0 + 16 + i) * HH + col] = (__bf16)v1;
        }
    }
}

// -------------------------------------------------------------------
// K3: global_add_pool: pooled[batch[i]] += act[i]  (bf16 -> f32 atomics)
// -------------------------------------------------------------------
__global__ void gnn_pool_add(const __bf16* __restrict__ act,
                             const int* __restrict__ batch,
                             float* __restrict__ pooled)
{
    int tid = blockIdx.x * 256 + threadIdx.x;
    int i   = tid >> 5;
    int c   = (tid & 31) << 3;
    if (i >= NN) return;
    int gidx = batch[i];
    union { uint4 q; __bf16 h[8]; } v;
    v.q = *(const uint4*)(act + (size_t)i * HH + c);
    float* o = pooled + (size_t)gidx * HH + c;
    #pragma unroll
    for (int j = 0; j < 8; ++j)
        atomicAdd(o + j, (float)v.h[j]);
}

// -------------------------------------------------------------------
// K4: out[g] = pooled[g] . Wpost + bpost   (one block per graph)
// -------------------------------------------------------------------
__global__ void gnn_final_linear(const float* __restrict__ pooled,
                                 const float* __restrict__ Wpost,
                                 const float* __restrict__ bpost,
                                 float* __restrict__ out)
{
    __shared__ float red[256];
    const int gidx = blockIdx.x;
    const int t    = threadIdx.x;
    red[t] = pooled[(size_t)gidx * HH + t] * Wpost[t];
    __syncthreads();
    for (int s = 128; s > 0; s >>= 1) {
        if (t < s) red[t] += red[t + s];
        __syncthreads();
    }
    if (t == 0) out[gidx] = red[0] + bpost[0];
}

// -------------------------------------------------------------------
// Host launcher (graph-capture safe: only async ops on `stream`)
// -------------------------------------------------------------------
extern "C" void kernel_launch(void* const* d_in, const int* in_sizes, int n_in,
                              void* d_out, int out_size, void* d_ws, size_t ws_size,
                              hipStream_t stream)
{
    const float* x      = (const float*)d_in[0];
    const int*   ei     = (const int*)  d_in[1];   // [2*E]: src then dst
    const int*   batch  = (const int*)  d_in[2];
    const float* Wrel1  = (const float*)d_in[3];
    const float* brel1  = (const float*)d_in[4];
    const float* Wroot1 = (const float*)d_in[5];
    const float* a1     = (const float*)d_in[6];
    const float* Wrel2  = (const float*)d_in[7];
    const float* brel2  = (const float*)d_in[8];
    const float* Wroot2 = (const float*)d_in[9];
    const float* a2     = (const float*)d_in[10];
    const float* Wpost  = (const float*)d_in[11];
    const float* bpost  = (const float*)d_in[12];

    // workspace layout (offsets 256B aligned)
    char* ws = (char*)d_ws;
    const size_t AGG_BYTES = (size_t)NN * HH * sizeof(float);   // 102,400,000
    const size_t ACT_BYTES = (size_t)NN * HH * sizeof(__bf16);  //  51,200,000
    const size_t WSZ_BYTES = (size_t)KTOT * HH * sizeof(__bf16);//     262,144
    float*  agg    = (float*)(ws);
    __bf16* act    = (__bf16*)(ws + AGG_BYTES);                 // x -> h1 -> h2
    __bf16* Wswz1  = (__bf16*)(ws + AGG_BYTES + ACT_BYTES);
    __bf16* Wswz2  = (__bf16*)(ws + AGG_BYTES + ACT_BYTES + WSZ_BYTES);
    float*  pooled = (float*)(ws + AGG_BYTES + ACT_BYTES + 2 * WSZ_BYTES);

    hipMemsetAsync(pooled, 0, (size_t)GG * HH * sizeof(float), stream);

    // weight conversion + swizzle (512x256 each)
    gnn_swizzle_weights<<<KTOT * HH / 256, 256, 0, stream>>>(Wrel1, Wroot1, Wswz1);
    gnn_swizzle_weights<<<KTOT * HH / 256, 256, 0, stream>>>(Wrel2, Wroot2, Wswz2);

    // act = bf16(x)
    gnn_cvt_bf16<<<(NN * HH / 8) / 256, 256, 0, stream>>>(x, act);

    // layer 1
    hipMemsetAsync(agg, 0, AGG_BYTES, stream);
    gnn_scatter_add<<<EE * 32 / 256, 256, 0, stream>>>(act, ei, ei + EE, agg);
    gnn_gemm_bias_prelu<<<NN / 32, 64, 0, stream>>>(agg, act, brel1, a1, Wswz1, act);

    // layer 2 (agg re-zeroed and reused)
    hipMemsetAsync(agg, 0, AGG_BYTES, stream);
    gnn_scatter_add<<<EE * 32 / 256, 256, 0, stream>>>(act, ei, ei + EE, agg);
    gnn_gemm_bias_prelu<<<NN / 32, 64, 0, stream>>>(agg, act, brel2, a2, Wswz2, act);

    // pool + final linear
    gnn_pool_add<<<NN * 32 / 256, 256, 0, stream>>>(act, batch, pooled);
    gnn_final_linear<<<GG, 256, 0, stream>>>(pooled, Wpost, bpost, (float*)d_out);
}
